// LocalTransformerBlock_49022756716966
// MI455X (gfx1250) — matrix operand order repaired and checked by reading, not compiled, not run on hardware
//
#include <hip/hip_runtime.h>
#include <hip/hip_fp16.h>

typedef _Float16 v16h __attribute__((ext_vector_type(16)));
typedef _Float16 v8h  __attribute__((ext_vector_type(8)));
typedef float    v8f  __attribute__((ext_vector_type(8)));

#define BATCH 64
#define HH    56
#define WW2   56
#define C_    192
#define C3    576
#define HEADS 6
#define HD    32
#define WIN   7
#define SHIFT 3
#define NTOK  49
#define NPAD  64
#define NWIN  4096          // BATCH * 64 windows
#define EPS   1e-5f
#define SCALE 0.17677669529663687f   // 1/sqrt(32)

// ---------------- fragment loaders (contiguous k order, same for A and B) ----------------

// A fragment 16x32 f16, source row-major with leading dim ld (halves).
// lane L: m = L&15 ; element e: k = 16*(L>>4) + e
__device__ __forceinline__ v16h load_A(const _Float16* base, int ld, int lane) {
    const _Float16* p = base + (lane & 15) * ld + ((lane >> 4) << 4);
    v8h a0 = *(const v8h*)(p);
    v8h a1 = *(const v8h*)(p + 8);
    v16h r;
#pragma unroll
    for (int i = 0; i < 8; ++i) { r[i] = a0[i]; r[8 + i] = a1[i]; }
    return r;
}

// B fragment 32x16 f16 where B[k][n] = M[n][k] (M row-major N x K, ld in halves).
// lane L: n = L&15 ; element e: k = e + 16*(L>>4)   (contiguous in M's row)
__device__ __forceinline__ v16h load_Bt(const _Float16* base, int ld, int lane) {
    const _Float16* p = base + (lane & 15) * ld + ((lane >> 4) << 4);
    v8h b0 = *(const v8h*)(p);
    v8h b1 = *(const v8h*)(p + 8);
    v16h r;
#pragma unroll
    for (int i = 0; i < 8; ++i) { r[i] = b0[i]; r[8 + i] = b1[i]; }
    return r;
}

// B fragment 32x16 f16 where B[k][n] = M[k][n] (M row-major K x N, ld in halves).
// lane L: n = L&15 ; element e: k = e + 16*(L>>4)   (strided by ld)
__device__ __forceinline__ v16h load_Bn(const _Float16* base, int ld, int lane) {
    const _Float16* p = base + ((lane >> 4) << 4) * ld + (lane & 15);
    v16h r;
#pragma unroll
    for (int e = 0; e < 16; ++e) r[e] = p[e * ld];
    return r;
}

__device__ __forceinline__ v8f wmma_f16(v16h a, v16h b, v8f c) {
    return __builtin_amdgcn_wmma_f32_16x16x32_f16(false, a, false, b, (short)0, c,
                                                  false, false);
}

__device__ __forceinline__ v8f vzero8() {
    v8f z;
#pragma unroll
    for (int i = 0; i < 8; ++i) z[i] = 0.0f;
    return z;
}

// ------ weight prep: fp32 -> f16, transposed to [out][in]; Q columns pre-scaled ------

__global__ void prep_weights(const float* __restrict__ w_qkv,   // [192][576]
                             const float* __restrict__ w_proj,  // [192][192]
                             _Float16* __restrict__ WqkvT,      // [576][192]
                             _Float16* __restrict__ WprojT) {   // [192][192]
    int i = blockIdx.x * 256 + threadIdx.x;
    if (i < C_ * C3) {
        int k = i / C3, o = i % C3;
        float v = w_qkv[i];
        if (o < C_) v *= SCALE;            // fold attention scale into Q weights
        WqkvT[o * C_ + k] = (_Float16)v;
    }
    if (i < C_ * C_) {
        int k = i / C_, o = i % C_;
        WprojT[o * C_ + k] = (_Float16)w_proj[i];
    }
}

// ---------------- fully fused Swin block: one workgroup per window ----------------

__global__ __launch_bounds__(256) void
swin_block_kernel(const float* __restrict__ x,
                  const float* __restrict__ gamma,
                  const float* __restrict__ beta,
                  const float* __restrict__ b_qkv,
                  const float* __restrict__ rel_table,   // [169][6]
                  const float* __restrict__ b_proj,
                  const float* __restrict__ mask_matrix, // [64][49][49]
                  const _Float16* __restrict__ WqkvT,    // [576][192]
                  const _Float16* __restrict__ WprojT,   // [192][192]
                  float* __restrict__ out) {
    __shared__ __align__(16) _Float16 Xs[NPAD * C_];            // 24 KB, reused as O
    __shared__ __align__(16) _Float16 QKVs[NPAD * C3];          // 72 KB (Q|K|V)
    __shared__ __align__(16) _Float16 Ssm[HEADS * NPAD * NPAD]; // 48 KB logits/probs

    const int wi   = blockIdx.x;
    const int img  = wi >> 6;
    const int wIdx = wi & 63;
    const int wh   = wIdx >> 3;
    const int ww   = wIdx & 7;
    const int lane = threadIdx.x & 31;
    const int wave = threadIdx.x >> 5;

    // warm L2/near caches for the shared weights
    for (int i = threadIdx.x; i < 864; i += 256)
        __builtin_prefetch(((const char*)WqkvT) + i * 256, 0, 1);

    // ---- phase 1: LayerNorm + cyclic shift + window partition -> Xs (f16) ----
    for (int n = wave; n < NTOK; n += 8) {
        int rr = n / WIN, cc = n - rr * WIN;
        int hs = (wh * WIN + rr + SHIFT) % HH;
        int ws = (ww * WIN + cc + SHIFT) % WW2;
        const float* px = x + ((size_t)((img * HH + hs) * WW2 + ws)) * C_;
        float v[6], s = 0.0f, ss = 0.0f;
#pragma unroll
        for (int j = 0; j < 6; ++j) {
            v[j] = px[j * 32 + lane];
            s += v[j];
            ss += v[j] * v[j];
        }
#pragma unroll
        for (int m = 16; m >= 1; m >>= 1) {
            s  += __shfl_xor(s, m, 32);
            ss += __shfl_xor(ss, m, 32);
        }
        float mu  = s * (1.0f / C_);
        float var = ss * (1.0f / C_) - mu * mu;
        float inv = rsqrtf(var + EPS);
#pragma unroll
        for (int j = 0; j < 6; ++j) {
            int c = j * 32 + lane;
            Xs[n * C_ + c] = (_Float16)((v[j] - mu) * inv * gamma[c] + beta[c]);
        }
    }
    for (int i = threadIdx.x; i < (NPAD - NTOK) * C_; i += 256)
        Xs[NTOK * C_ + i] = (_Float16)0.0f;
    __syncthreads();

    // ---- phase 2: QKV GEMM  (64x192 @ 192x576) -> QKVs (f16) ----
    for (int t = wave; t < 4 * 36; t += 8) {
        int tm = t / 36, tn = t % 36;
        v8f acc = vzero8();
#pragma unroll
        for (int kk = 0; kk < C_; kk += 32) {
            v16h a = load_A(Xs + tm * 16 * C_ + kk, C_, lane);
            v16h b = load_Bt(WqkvT + (tn * 16) * C_ + kk, C_, lane);
            acc = wmma_f16(a, b, acc);
        }
        int col = tn * 16 + (lane & 15);
        float bb = b_qkv[col];
        if (col < C_) bb *= SCALE;         // match pre-scaled Q weights
#pragma unroll
        for (int r = 0; r < 8; ++r) {
            int row = tm * 16 + r + ((lane >> 4) << 3);
            QKVs[row * C3 + col] = (_Float16)(acc[r] + bb);
        }
    }
    __syncthreads();

    // ---- phase 3: S = Q*K^T + relpos + mask (96 tiles over all 8 waves) ----
    {
        const float* maskp = mask_matrix + wIdx * NTOK * NTOK;
        for (int t = wave; t < HEADS * 16; t += 8) {
            int h  = t >> 4;
            int tt = t & 15;
            int tm = tt >> 2, tn = tt & 3;
            const _Float16* Qh = QKVs + h * HD;
            const _Float16* Kh = QKVs + C_ + h * HD;
            v16h a = load_A(Qh + tm * 16 * C3, C3, lane);
            v16h b = load_Bt(Kh + tn * 16 * C3, C3, lane);
            v8f acc = vzero8();
            acc = wmma_f16(a, b, acc);
            int n = tn * 16 + (lane & 15);
#pragma unroll
            for (int r = 0; r < 8; ++r) {
                int m = tm * 16 + r + ((lane >> 4) << 3);
                float vv;
                if (m < NTOK && n < NTOK) {
                    int rpi = (m / WIN - n / WIN + WIN - 1) * (2 * WIN - 1) +
                              (m % WIN - n % WIN + WIN - 1);
                    vv = acc[r] + rel_table[rpi * HEADS + h] + maskp[m * NTOK + n];
                } else {
                    vv = -3.0e38f;  // -> f16 -inf; padding masked out
                }
                Ssm[h * (NPAD * NPAD) + m * NPAD + n] = (_Float16)vv;
            }
        }
    }
    __syncthreads();

    // ---- phase 4: row softmax over 49 valid columns (384 rows over 256 threads) ----
    for (int i = threadIdx.x; i < HEADS * NPAD; i += 256) {
        int h = i >> 6, r = i & 63;
        _Float16* row = Ssm + h * (NPAD * NPAD) + r * NPAD;
        if (r >= NTOK) {
            for (int c = 0; c < NPAD; ++c) row[c] = (_Float16)0.0f;
            continue;
        }
        float mx = -3.0e38f;
        for (int c = 0; c < NTOK; ++c) mx = fmaxf(mx, (float)row[c]);
        float sum = 0.0f;
        for (int c = 0; c < NTOK; ++c) {
            float e = __expf((float)row[c] - mx);
            row[c] = (_Float16)e;
            sum += e;
        }
        float invs = 1.0f / sum;
        for (int c = 0; c < NTOK; ++c)
            row[c] = (_Float16)((float)row[c] * invs);
        for (int c = NTOK; c < NPAD; ++c) row[c] = (_Float16)0.0f;
    }
    __syncthreads();

    // ---- phase 5: O = P @ V -> Xs reused as O (48 tiles over all 8 waves) ----
    for (int t = wave; t < HEADS * 8; t += 8) {
        int h  = t >> 3;
        int tt = t & 7;
        int tm = tt >> 1, tn = tt & 1;
        const _Float16* Ph = Ssm + h * (NPAD * NPAD);
        const _Float16* Vh = QKVs + 2 * C_ + h * HD;
        v8f acc = vzero8();
#pragma unroll
        for (int kk = 0; kk < NPAD; kk += 32) {
            v16h a = load_A(Ph + tm * 16 * NPAD + kk, NPAD, lane);
            v16h b = load_Bn(Vh + kk * C3 + tn * 16, C3, lane);
            acc = wmma_f16(a, b, acc);
        }
        int d = tn * 16 + (lane & 15);
#pragma unroll
        for (int r = 0; r < 8; ++r) {
            int row = tm * 16 + r + ((lane >> 4) << 3);
            Xs[row * C_ + h * HD + d] = (_Float16)acc[r];
        }
    }
    __syncthreads();

    // ---- phase 6: projection + window reverse + reverse shift -> out (f32) ----
    for (int t = wave; t < 4 * 12; t += 8) {
        int tm = t / 12, tn = t % 12;
        v8f acc = vzero8();
#pragma unroll
        for (int kk = 0; kk < C_; kk += 32) {
            v16h a = load_A(Xs + tm * 16 * C_ + kk, C_, lane);
            v16h b = load_Bt(WprojT + (tn * 16) * C_ + kk, C_, lane);
            acc = wmma_f16(a, b, acc);
        }
        int c = tn * 16 + (lane & 15);
        float bp = b_proj[c];
#pragma unroll
        for (int r = 0; r < 8; ++r) {
            int row = tm * 16 + r + ((lane >> 4) << 3);
            if (row < NTOK) {
                int rr = row / WIN, cc = row - rr * WIN;
                int hd2 = (wh * WIN + rr + SHIFT) % HH;
                int wd2 = (ww * WIN + cc + SHIFT) % WW2;
                out[((size_t)((img * HH + hd2) * WW2 + wd2)) * C_ + c] = acc[r] + bp;
            }
        }
    }
}

// ---------------- host launcher ----------------

extern "C" void kernel_launch(void* const* d_in, const int* in_sizes, int n_in,
                              void* d_out, int out_size, void* d_ws, size_t ws_size,
                              hipStream_t stream) {
    (void)in_sizes; (void)n_in; (void)out_size; (void)ws_size;
    const float* x           = (const float*)d_in[0];
    const float* gamma       = (const float*)d_in[1];
    const float* beta        = (const float*)d_in[2];
    const float* w_qkv       = (const float*)d_in[3];
    const float* b_qkv       = (const float*)d_in[4];
    const float* rel_table   = (const float*)d_in[5];
    const float* w_proj      = (const float*)d_in[6];
    const float* b_proj      = (const float*)d_in[7];
    const float* mask_matrix = (const float*)d_in[8];
    float* out = (float*)d_out;

    _Float16* WqkvT  = (_Float16*)d_ws;                 // 576*192 f16
    _Float16* WprojT = WqkvT + C3 * C_;                 // 192*192 f16

    prep_weights<<<(C_ * C3 + 255) / 256, 256, 0, stream>>>(w_qkv, w_proj, WqkvT, WprojT);
    swin_block_kernel<<<NWIN, 256, 0, stream>>>(x, gamma, beta, b_qkv, rel_table,
                                                b_proj, mask_matrix, WqkvT, WprojT, out);
}
